// CellLSTM_70738111365196
// MI455X (gfx1250) — matrix operand order repaired and checked by reading, not compiled, not run on hardware
//
#include <hip/hip_runtime.h>
#include <hip/hip_bf16.h>

// ---------------------------------------------------------------------------
// LSTM cell, B=131072, H=I=128.
// Fused GEMM: gates[b,n] = [x|h0][b, 0:256] . Wcat[0:256, n] + bias[n], n in [0,512)
// 34.4 GFLOP vs 576 MB HBM traffic -> ~25us memory floor @ 23.3 TB/s.
// Needs >=1.4 PF/s matrix throughput to stay memory bound -> bf16 WMMA
// (v_wmma_f32_16x16x32_bf16, f32 accumulate). Weights = 256 KB bf16, L2-resident.
// ---------------------------------------------------------------------------

typedef __bf16 bf16_t;
typedef __attribute__((ext_vector_type(16))) __bf16 bf16x16;
typedef __attribute__((ext_vector_type(8)))  __bf16 bf16x8;
typedef __attribute__((ext_vector_type(8)))  float  f32x8;
typedef __attribute__((ext_vector_type(4)))  float  f32x4;

#define B_TOTAL   131072
#define HDIM      128
#define NCOLS     512          // 4 gates * 128
#define KDIM      256          // 128 (x) + 128 (h0)
#define BM        128          // batch rows per block
#define LDS_PAD   8
#define LDS_STRIDE (KDIM + LDS_PAD)   // bf16 elements per LDS row
#define WPACK_ELEMS (32*8*32*16)      // 32 col-tiles * 8 k-frags * 32 lanes * 16 vals

__device__ __forceinline__ float sigm(float x) {
    return 1.0f / (1.0f + __expf(-x));
}
__device__ __forceinline__ float tanh_fast(float x) {
    // tanh(x) = 1 - 2/(e^{2x}+1); saturates correctly at +-1 for large |x|
    float e = __expf(2.0f * x);
    return 1.0f - 2.0f / (e + 1.0f);
}

// ---------------------------------------------------------------------------
// Pack kernel: build bf16 weights in WMMA B-fragment layout + fused bias.
// Wcat[k, n]: k<128 -> ih[(g*128+k)*128 + i], k>=128 -> hh[(g*128+(k-128))*128+i]
//             with n = g*128 + i  (gates[g] = x @ ih4[g], row-of-weight = h index)
// Fragment layout (B = 32x16, 16-bit): lanes 0-15 hold K=kf*32+{0..15},
// lanes 16-31 hold K=kf*32+{16..31}; N = ct*16 + (lane&15). Each lane's 16
// bf16 values stored contiguously -> two global_load_b128 per fragment.
// ---------------------------------------------------------------------------
__global__ void lstm_pack(const float* __restrict__ ih, const float* __restrict__ hh,
                          const float* __restrict__ ib, const float* __restrict__ hb,
                          bf16_t* __restrict__ wpack, float* __restrict__ bias)
{
    int idx = blockIdx.x * blockDim.x + threadIdx.x;
    if (idx < WPACK_ELEMS) {
        int e    =  idx        & 15;
        int lane = (idx >> 4)  & 31;
        int kf   = (idx >> 9)  & 7;
        int ct   =  idx >> 12;             // 0..31
        int K = kf * 32 + ((lane >> 4) << 4) + e;   // lanes 16-31 -> +16
        int N = ct * 16 + (lane & 15);
        int g = N >> 7;
        int i = N & 127;
        int h = K & 127;
        float v = (K < 128) ? ih[(g * 128 + h) * 128 + i]
                            : hh[(g * 128 + h) * 128 + i];
        wpack[idx] = (bf16_t)v;
    }
    if (idx < NCOLS) bias[idx] = ib[idx] + hb[idx];
}

// ---------------------------------------------------------------------------
// Main kernel: 1024 blocks x 256 threads (8 wave32). Block = 128 batch rows.
// ---------------------------------------------------------------------------
__global__ __launch_bounds__(256) void lstm_gemm(
    const float*  __restrict__ x,
    const float*  __restrict__ h0,
    const float*  __restrict__ c0,
    const bf16_t* __restrict__ wpack,
    const float*  __restrict__ bias,
    float*        __restrict__ out)
{
    __shared__ bf16_t a_lds[BM][LDS_STRIDE];

    const int t = threadIdx.x;
    const int blockRow = blockIdx.x * BM;

    // ---- Stage [x | h0] tile into LDS as bf16 (fp32 demote in flight) ----
    // 128 rows x 32 float4 per source matrix = 4096 f32x4 / 256 threads = 16 ea.
    #pragma unroll
    for (int c = 0; c < 16; ++c) {
        int id   = c * 256 + t;
        int row  = id >> 5;            // /32
        int col4 = (id & 31) * 4;
        size_t goff = (size_t)(blockRow + row) * HDIM + col4;
        f32x4 vx = *(const f32x4*)(x  + goff);
        f32x4 vh = *(const f32x4*)(h0 + goff);
        bf16_t* px = &a_lds[row][col4];
        px[0] = (bf16_t)vx.x; px[1] = (bf16_t)vx.y;
        px[2] = (bf16_t)vx.z; px[3] = (bf16_t)vx.w;
        bf16_t* ph = &a_lds[row][HDIM + col4];
        ph[0] = (bf16_t)vh.x; ph[1] = (bf16_t)vh.y;
        ph[2] = (bf16_t)vh.z; ph[3] = (bf16_t)vh.w;
    }
    __syncthreads();

    const int wave   = t >> 5;
    const int lane   = t & 31;
    const int lane16 = lane & 15;
    const bool hi    = (lane >= 16);
    const int arow   = wave * 16 + lane16;   // lanes l and l+16 share row M=l

    // ---- Preload all 8 A fragments (16x32 bf16 each) for this 16-row strip.
    // A-layout: lanes 0-15 -> K = kf*32 + {0..15};
    //           lanes 16-31 -> K = kf*32 + {16..31}.
    bf16x16 afrag[8];
    #pragma unroll
    for (int kf = 0; kf < 8; ++kf) {
        int k0 = kf * 32 + (hi ? 16 : 0);
        union { bf16x16 v; bf16x8 h[2]; } u;
        u.h[0] = *(const bf16x8*)&a_lds[arow][k0];
        u.h[1] = *(const bf16x8*)&a_lds[arow][k0 + 8];
        afrag[kf] = u.v;
    }

    const size_t BH = (size_t)B_TOTAL * HDIM;
    float* outH = out;
    float* outC = out + BH;
    float* outI = out + 2 * BH;
    float* outF = out + 3 * BH;
    float* outG = out + 4 * BH;
    float* outO = out + 5 * BH;

    // ---- 8 i-tiles of 16 hidden columns; each needs 4 gate accumulators ----
    for (int it = 0; it < 8; ++it) {
        f32x8 acc[4];
        #pragma unroll
        for (int g = 0; g < 4; ++g) {
            // C/D layout: all 8 acc elements of a lane share column N=lane16.
            float bv = bias[g * 128 + it * 16 + lane16];
            f32x8 a = {bv, bv, bv, bv, bv, bv, bv, bv};
            int ct = g * 8 + it;   // global column tile (n base = ct*16)
            #pragma unroll
            for (int kf = 0; kf < 8; ++kf) {
                const bf16_t* p = wpack + (((size_t)(ct * 8 + kf) * 32 + lane) << 4);
                union { bf16x16 v; bf16x8 h[2]; } bu;
                bu.h[0] = *(const bf16x8*)p;        // global_load_b128
                bu.h[1] = *(const bf16x8*)(p + 8);  // global_load_b128
                a = __builtin_amdgcn_wmma_f32_16x16x32_bf16(
                        /*neg_a=*/false, afrag[kf],
                        /*neg_b=*/false, bu.v,
                        /*c_mod=*/(short)0, a,
                        /*reuse_a=*/false, /*reuse_b=*/false);
            }
            acc[g] = a;
        }

        // ---- Fused LSTM elementwise + 6 stores ----
        // C/D layout: VGPR j -> row M = j + (lane>=16 ? 8 : 0), col N = lane16.
        int colI = it * 16 + lane16;
        #pragma unroll
        for (int j = 0; j < 8; ++j) {
            int M = j + (hi ? 8 : 0);
            size_t b   = (size_t)blockRow + wave * 16 + M;
            size_t off = b * HDIM + colI;
            float Ig = acc[0][j], Fg = acc[1][j], Gg = acc[2][j], Og = acc[3][j];
            float c0v = c0[off];
            float c1 = c0v * sigm(Fg) + sigm(Ig) * tanh_fast(Gg);
            float h1 = sigm(Og) + tanh_fast(c1);   // faithful: additive, not gated
            outH[off] = h1;
            outC[off] = c1;
            outI[off] = Ig;
            outF[off] = Fg;
            outG[off] = Gg;
            outO[off] = Og;
        }
    }
}

// ---------------------------------------------------------------------------
extern "C" void kernel_launch(void* const* d_in, const int* in_sizes, int n_in,
                              void* d_out, int out_size, void* d_ws, size_t ws_size,
                              hipStream_t stream)
{
    const float* x  = (const float*)d_in[0];
    const float* h0 = (const float*)d_in[1];
    const float* c0 = (const float*)d_in[2];
    const float* ih = (const float*)d_in[3];
    const float* hh = (const float*)d_in[4];
    const float* ib = (const float*)d_in[5];
    const float* hb = (const float*)d_in[6];

    bf16_t* wpack = (bf16_t*)d_ws;                                   // 256 KB
    float*  bias  = (float*)((char*)d_ws + WPACK_ELEMS * sizeof(bf16_t)); // 2 KB

    // Pack weights into WMMA fragment layout + fuse biases (tiny, L2-resident).
    lstm_pack<<<WPACK_ELEMS / 256, 256, 0, stream>>>(ih, hh, ib, hb, wpack, bias);

    // Main fused GEMM + LSTM elementwise.
    lstm_gemm<<<B_TOTAL / BM, 256, 0, stream>>>(x, h0, c0, wpack, bias, (float*)d_out);
}
